// MultiHeadCrossAttention_15281493639604
// MI455X (gfx1250) — matrix operand order repaired and checked by reading, not compiled, not run on hardware
//
#include <hip/hip_runtime.h>
#include <hip/hip_bf16.h>

// ---------------------------------------------------------------------------
// MultiHeadCrossAttention on gfx1250 (wave32, WMMA bf16 -> f32 accumulate)
// B=2, QL=2048, KVL=4096, HIDDEN=1024, 16 heads x d=64, scale=1/8
// Flash attention stages K/V tiles into LDS with the Tensor Data Mover
// (tensor_load_to_lds + s_wait_tensorcnt), shared by all 8 waves of a block.
// ---------------------------------------------------------------------------

#define NH      16
#define HD      64
#define HIDDEN_ 1024
#define BATCH_  2
#define QLEN_   2048
#define KVLEN_  4096

typedef __attribute__((ext_vector_type(16))) __bf16 v16bf;
typedef __attribute__((ext_vector_type(8)))  float  v8f;
typedef __attribute__((ext_vector_type(4)))  unsigned v4u;
typedef __attribute__((ext_vector_type(8)))  int      v8i;
typedef __attribute__((ext_vector_type(4)))  int      v4i;

// Native conversion -> v_cvt_*bf16_f32 (RNE)
__device__ __forceinline__ unsigned short bf16_bits(float f) {
  union { __bf16 h; unsigned short s; } u;
  u.h = (__bf16)f;
  return u.s;
}

union Frag {
  v16bf v;
  uint4 u4[2];
  unsigned u[8];
  unsigned short s[16];
};

__device__ __forceinline__ v8f zero8() {
  v8f z = {0.f, 0.f, 0.f, 0.f, 0.f, 0.f, 0.f, 0.f};
  return z;
}

__device__ __forceinline__ v8f wmma_bf16(v16bf a, v16bf b, v8f c) {
  return __builtin_amdgcn_wmma_f32_16x16x32_bf16(false, a, false, b, (short)0, c,
                                                 false, false);
}

// A-matrix 16x32 fragment from a row-major bf16 row (lane holds row m=lane%16).
// element e -> k = k0 + 16*hf + e
__device__ __forceinline__ Frag load_a_bf16(const unsigned short* row, int k0, int hf) {
  Frag f;
  f.u4[0] = *(const uint4*)(row + k0 + 16 * hf);
  f.u4[1] = *(const uint4*)(row + k0 + 16 * hf + 8);
  return f;
}
// B-matrix 32x16 fragment from a row-major bf16 row (lane holds col n=lane%16).
// element e -> k = k0 + 16*hf + e (contiguous k, same order as load_a_bf16)
__device__ __forceinline__ Frag load_b_bf16(const unsigned short* row, int k0, int hf) {
  Frag f;
  f.u4[0] = *(const uint4*)(row + k0 + 16 * hf);
  f.u4[1] = *(const uint4*)(row + k0 + 16 * hf + 8);
  return f;
}

// ---------------------------------------------------------------------------
// TDM 2D tile load: global (data_size=2B) -> LDS, with per-row LDS padding.
// D# bitfields per CDNA5 ISA 8.3/8.4 (group0 128b, group1 256b, groups2/3 = 0).
// cfg packs: data_size=1(2B) | pad_enable | pad_interval | pad_amount.
// ---------------------------------------------------------------------------
__device__ __forceinline__ void tdm_load_2d(unsigned lds_addr, unsigned long long gaddr,
                                            unsigned cfg, unsigned dim0, unsigned dim1,
                                            unsigned tile0, unsigned tile1,
                                            unsigned stride0) {
  v4u g0;
  g0[0] = 1u;                                   // count=1, user mode, no gather
  g0[1] = lds_addr;                             // lds_addr [63:32]
  g0[2] = (unsigned)gaddr;                      // global_addr [95:64]
  g0[3] = ((unsigned)(gaddr >> 32) & 0x01FFFFFFu) | (2u << 30);  // [120:96], type=2
  v8i g1;
  g1[0] = (int)cfg;                             // mask=0 | data_size | pad ctrl
  g1[1] = (int)(dim0 << 16);                    // tensor_dim0 [79:48]
  g1[2] = (int)((dim0 >> 16) | (dim1 << 16));   // tensor_dim1 [111:80]
  g1[3] = (int)((dim1 >> 16) | (tile0 << 16));  // tile_dim0 [127:112]
  g1[4] = (int)tile1;                           // tile_dim1 [143:128], tile_dim2=0
  g1[5] = (int)stride0;                         // tensor_dim0_stride [207:160]
  g1[6] = 0;
  g1[7] = 0;
  v4i gz = {0, 0, 0, 0};
#if __clang_major__ >= 23
  v8i gz8 = {0, 0, 0, 0, 0, 0, 0, 0};
  __builtin_amdgcn_tensor_load_to_lds(g0, g1, gz, gz, gz8, 0);
#else
  __builtin_amdgcn_tensor_load_to_lds(g0, g1, gz, gz, 0);
#endif
}

// data_size=2B | pad_enable | pad_interval(128B rows) | pad_amount(16B)
#define TDM_CFG_K ((1u << 16) | (1u << 20) | (4u << 22) | (3u << 25))
// data_size=2B | pad_enable | pad_interval(64B rows)  | pad_amount(16B)
#define TDM_CFG_V ((1u << 16) | (1u << 20) | (3u << 22) | (3u << 25))

#define KT_PITCH 72   // 64 elems + 8 pad  (144B rows: 16-lane reads conflict-free)
#define VT_PITCH 40   // 32 elems + 8 pad  ( 80B rows)

// ---------------------------------------------------------------------------
// Elementwise f32 -> bf16 staging (one pass; GEMM loops then do zero converts).
// ---------------------------------------------------------------------------
__global__ __launch_bounds__(256) void cvt_kernel(const float* __restrict__ src,
                                                  unsigned short* __restrict__ dst,
                                                  int n8) {
  int t = (int)(blockIdx.x * blockDim.x + threadIdx.x);
  if (t >= n8) return;
  size_t i = (size_t)t * 8;
  float4 a = *(const float4*)(src + i);
  float4 b = *(const float4*)(src + i + 4);
  Frag f;
  f.s[0] = bf16_bits(a.x); f.s[1] = bf16_bits(a.y);
  f.s[2] = bf16_bits(a.z); f.s[3] = bf16_bits(a.w);
  f.s[4] = bf16_bits(b.x); f.s[5] = bf16_bits(b.y);
  f.s[6] = bf16_bits(b.z); f.s[7] = bf16_bits(b.w);
  *(uint4*)(dst + i) = f.u4[0];
}

// ---------------------------------------------------------------------------
// Projection GEMM:  Y = (X @ W^T) * scale   (bf16 in, bf16 out, f32 accum)
// One wave computes a 16x64 output tile (4 WMMA accumulators, A-frag reuse x4).
// layout 0: [b, h, s, d]   (Q, K)     layout 1: [b, h, d, s]   (V transposed)
// ---------------------------------------------------------------------------
__global__ __launch_bounds__(256) void proj_kernel(
    const unsigned short* __restrict__ X, const unsigned short* __restrict__ W,
    unsigned short* __restrict__ Y, int Lshift, int layout, float scale) {
  int wid = (int)((blockIdx.x * blockDim.x + threadIdx.x) >> 5);
  int lane = (int)(threadIdx.x & 31);
  int nl = lane & 15, hf = lane >> 4;
  int mt = wid >> 4;
  int nt = wid & 15;
  int m0 = mt * 16, n0 = nt * 64;

  const unsigned short* xrow = X + (size_t)(m0 + nl) * HIDDEN_;
  v8f acc[4] = {zero8(), zero8(), zero8(), zero8()};

  for (int k0 = 0; k0 < HIDDEN_; k0 += 32) {
    Frag a = load_a_bf16(xrow, k0, hf);
#pragma unroll
    for (int j = 0; j < 4; ++j) {
      const unsigned short* wrow = W + (size_t)(n0 + j * 16 + nl) * HIDDEN_;
      Frag b = load_b_bf16(wrow, k0, hf);
      acc[j] = wmma_bf16(a.v, b.v, acc[j]);
    }
  }

  int Lmask = (1 << Lshift) - 1;
#pragma unroll
  for (int j = 0; j < 4; ++j) {
#pragma unroll
    for (int r = 0; r < 8; ++r) {
      int m = m0 + r + 8 * hf;
      int nn = n0 + j * 16 + nl;
      int bb = m >> Lshift, s = m & Lmask;
      int h = nn >> 6, d = nn & 63;
      size_t idx = (layout == 0)
          ? (((size_t)(bb * NH + h) << Lshift) + s) * HD + d
          : (((size_t)(bb * NH + h) * HD + d) << Lshift) + s;
      Y[idx] = bf16_bits(acc[j][r] * scale);
    }
  }
}

// ---------------------------------------------------------------------------
// Flash attention. Block = 8 waves = 8 query tiles of one (b,h); K/V tiles are
// TDM-staged into double-buffered LDS and shared by all waves. Each wave
// computes S^T = K*Q^T so the exp'd C-layout maps onto the B-fragment of
// O^T = V^T * P^T with a single half-wave shfl_xor(16) exchange.
// ---------------------------------------------------------------------------
__global__ __launch_bounds__(256) void flash_kernel(
    const unsigned short* __restrict__ Qh, const unsigned short* __restrict__ Kh,
    const unsigned short* __restrict__ Vt, unsigned short* __restrict__ Ao) {
  __shared__ alignas(16) unsigned short KT[2][32 * KT_PITCH];  // 9.0 KB
  __shared__ alignas(16) unsigned short VT[2][64 * VT_PITCH];  // 10.0 KB

  int tid = (int)threadIdx.x;
  int wave = tid >> 5;
  int lane = tid & 31;
  int nl = lane & 15, hf = lane >> 4;
  int bh = (int)blockIdx.x >> 4;              // 16 blocks per (b,h)
  int qt = ((int)blockIdx.x & 15) * 8 + wave; // this wave's query tile

  const unsigned short* Qb = Qh + (size_t)bh * QLEN_ * HD;
  const unsigned short* Kb = Kh + (size_t)bh * KVLEN_ * HD;
  const unsigned short* Vb = Vt + (size_t)bh * HD * KVLEN_;
  unsigned long long kbase = (unsigned long long)(size_t)Kb;
  unsigned long long vbase = (unsigned long long)(size_t)Vb;
  unsigned ldsK0 = (unsigned)(size_t)&KT[0][0];
  unsigned ldsK1 = (unsigned)(size_t)&KT[1][0];
  unsigned ldsV0 = (unsigned)(size_t)&VT[0][0];
  unsigned ldsV1 = (unsigned)(size_t)&VT[1][0];

  // Q^T B-fragments (column n = query), two d-halves of 32
  const unsigned short* qrow = Qb + (size_t)(qt * 16 + nl) * HD;
  Frag qf0 = load_b_bf16(qrow, 0, hf);
  Frag qf1 = load_b_bf16(qrow, 32, hf);

  v8f o[4] = {zero8(), zero8(), zero8(), zero8()};   // O^T, 64 x 16
  float mrun = -3.0e38f, lrun = 0.f;
  const float LOG2E = 1.4426950408889634f;
  const int NBLK = KVLEN_ / 32;   // 128 KV blocks

  if (wave == 0) {  // prologue: stage block 0 into buffer 0
    tdm_load_2d(ldsK0, kbase, TDM_CFG_K, HD, KVLEN_, HD, 32, HD);
    tdm_load_2d(ldsV0, vbase, TDM_CFG_V, KVLEN_, HD, 32, HD, KVLEN_);
  }

  for (int i = 0; i < NBLK; ++i) {
    int buf = i & 1;
    if (wave == 0) {
      if (i + 1 < NBLK) {   // prefetch next block into the other buffer
        unsigned long long kg = kbase + (unsigned long long)(i + 1) * 32 * HD * 2;
        unsigned long long vg = vbase + (unsigned long long)(i + 1) * 32 * 2;
        tdm_load_2d(buf ? ldsK0 : ldsK1, kg, TDM_CFG_K, HD, KVLEN_, HD, 32, HD);
        tdm_load_2d(buf ? ldsV0 : ldsV1, vg, TDM_CFG_V, KVLEN_, HD, 32, HD, KVLEN_);
        __builtin_amdgcn_s_wait_tensorcnt((short)2);  // block i landed
      } else {
        __builtin_amdgcn_s_wait_tensorcnt((short)0);
      }
    }
    __syncthreads();

    const unsigned short* KTb = KT[buf];
    const unsigned short* VTb = VT[buf];

    // S^T tiles from LDS (ds_load_b128 fragments, conflict-free pitches)
    const unsigned short* kr0 = KTb + nl * KT_PITCH;
    const unsigned short* kr1 = KTb + (16 + nl) * KT_PITCH;
    v8f s0 = zero8(), s1 = zero8();
    s0 = wmma_bf16(load_a_bf16(kr0, 0, hf).v,  qf0.v, s0);
    s0 = wmma_bf16(load_a_bf16(kr0, 32, hf).v, qf1.v, s0);
    s1 = wmma_bf16(load_a_bf16(kr1, 0, hf).v,  qf0.v, s1);
    s1 = wmma_bf16(load_a_bf16(kr1, 32, hf).v, qf1.v, s1);

    // online softmax over the 32 new scores of column q = nl
    float bm = s0[0];
#pragma unroll
    for (int r = 0; r < 8; ++r) { bm = fmaxf(bm, s0[r]); bm = fmaxf(bm, s1[r]); }
    bm = fmaxf(bm, __shfl_xor(bm, 16));
    float mn = fmaxf(mrun, bm);
    float alpha = exp2f((mrun - mn) * LOG2E);
    float p0[8], p1[8], lsum = 0.f;
#pragma unroll
    for (int r = 0; r < 8; ++r) {
      p0[r] = exp2f((s0[r] - mn) * LOG2E);
      p1[r] = exp2f((s1[r] - mn) * LOG2E);
      lsum += p0[r] + p1[r];
    }
    lsum += __shfl_xor(lsum, 16);
    mrun = mn;
    lrun = lrun * alpha + lsum;

    // Pack P to bf16 dwords; half-wave exchange builds the P^T B-fragment:
    //   lanes 0-15: e0..7 = own tile0, e8..15 = partner tile0
    //   lanes 16-31: e0..7 = partner tile1, e8..15 = own tile1
    unsigned u0[4], u1[4];
#pragma unroll
    for (int j = 0; j < 4; ++j) {
      u0[j] = (unsigned)bf16_bits(p0[2 * j]) | ((unsigned)bf16_bits(p0[2 * j + 1]) << 16);
      u1[j] = (unsigned)bf16_bits(p1[2 * j]) | ((unsigned)bf16_bits(p1[2 * j + 1]) << 16);
    }
    Frag pf;
#pragma unroll
    for (int j = 0; j < 4; ++j) {
      unsigned t0 = (unsigned)__shfl_xor((int)u0[j], 16);
      unsigned t1 = (unsigned)__shfl_xor((int)u1[j], 16);
      pf.u[j]     = hf ? t1 : u0[j];
      pf.u[j + 4] = hf ? u1[j] : t0;
    }

    // rescale running O and accumulate O^T += V^T * P^T (V fragments from LDS)
#pragma unroll
    for (int t = 0; t < 4; ++t) {
#pragma unroll
      for (int r = 0; r < 8; ++r) o[t][r] *= alpha;
    }
#pragma unroll
    for (int t = 0; t < 4; ++t) {
      const unsigned short* vr = VTb + (t * 16 + nl) * VT_PITCH;
      Frag vf = load_a_bf16(vr, 0, hf);
      o[t] = wmma_bf16(vf.v, pf.v, o[t]);
    }
    __syncthreads();   // all waves done with buf before it is re-staged
  }

  // normalize and store as [b, h, q, d] bf16 (8 consecutive d per 16B store)
  float inv = 1.0f / lrun;
  int q = qt * 16 + nl;
  unsigned short* orow = Ao + ((size_t)bh * QLEN_ + q) * HD;
#pragma unroll
  for (int t = 0; t < 4; ++t) {
    uint4 pk;
    pk.x = (unsigned)bf16_bits(o[t][0] * inv) | ((unsigned)bf16_bits(o[t][1] * inv) << 16);
    pk.y = (unsigned)bf16_bits(o[t][2] * inv) | ((unsigned)bf16_bits(o[t][3] * inv) << 16);
    pk.z = (unsigned)bf16_bits(o[t][4] * inv) | ((unsigned)bf16_bits(o[t][5] * inv) << 16);
    pk.w = (unsigned)bf16_bits(o[t][6] * inv) | ((unsigned)bf16_bits(o[t][7] * inv) << 16);
    *(uint4*)(orow + t * 16 + 8 * hf) = pk;
  }
}

// ---------------------------------------------------------------------------
// Output projection: out = concat_heads(Ao) @ Wo^T   (bf16 in, f32 out)
// ---------------------------------------------------------------------------
__global__ __launch_bounds__(256) void outproj_kernel(
    const unsigned short* __restrict__ Ao, const unsigned short* __restrict__ Wo,
    float* __restrict__ out) {
  int wid = (int)((blockIdx.x * blockDim.x + threadIdx.x) >> 5);
  int lane = (int)(threadIdx.x & 31);
  int nl = lane & 15, hf = lane >> 4;
  int mt = wid >> 4;
  int nt = wid & 15;
  int m0 = mt * 16, n0 = nt * 64;

  int m = m0 + nl;
  int bb = m >> 11, q = m & (QLEN_ - 1);
  v8f acc[4] = {zero8(), zero8(), zero8(), zero8()};

  for (int k0 = 0; k0 < HIDDEN_; k0 += 32) {
    int h = k0 >> 6;                  // 32-wide k block never crosses a head
    const unsigned short* arow = Ao + ((size_t)(bb * NH + h) * QLEN_ + q) * HD;
    Frag a;
    a.u4[0] = *(const uint4*)(arow + ((k0 + 16 * hf) & 63));
    a.u4[1] = *(const uint4*)(arow + ((k0 + 16 * hf + 8) & 63));
#pragma unroll
    for (int j = 0; j < 4; ++j) {
      const unsigned short* wrow = Wo + (size_t)(n0 + j * 16 + nl) * HIDDEN_;
      Frag b = load_b_bf16(wrow, k0, hf);
      acc[j] = wmma_bf16(a.v, b.v, acc[j]);
    }
  }

#pragma unroll
  for (int j = 0; j < 4; ++j) {
#pragma unroll
    for (int r = 0; r < 8; ++r)
      out[(size_t)(m0 + r + 8 * hf) * HIDDEN_ + n0 + j * 16 + nl] = acc[j][r];
  }
}

// ---------------------------------------------------------------------------
extern "C" void kernel_launch(void* const* d_in, const int* in_sizes, int n_in,
                              void* d_out, int out_size, void* d_ws, size_t ws_size,
                              hipStream_t stream) {
  (void)in_sizes; (void)n_in; (void)out_size; (void)ws_size;
  const float* q_in = (const float*)d_in[0];
  const float* k_in = (const float*)d_in[1];
  const float* v_in = (const float*)d_in[2];
  const float* Wq   = (const float*)d_in[3];
  const float* Wk   = (const float*)d_in[4];
  const float* Wv   = (const float*)d_in[5];
  const float* Wo   = (const float*)d_in[6];
  float* out = (float*)d_out;

  const size_t NQ = (size_t)BATCH_ * QLEN_ * HIDDEN_;    // 4 Mi elems
  const size_t NK = (size_t)BATCH_ * KVLEN_ * HIDDEN_;   // 8 Mi elems
  const size_t NW = (size_t)HIDDEN_ * HIDDEN_;           // 1 Mi elems

  // bf16 workspace layout (ushort elements):
  //  Qh | Kh | Vt | Wq|Wk|Wv|Wo | Xq | Xk | Xv ; Ao aliases Xq.  Total 88 MB.
  unsigned short* Qh  = (unsigned short*)d_ws;
  unsigned short* Kh  = Qh + NQ;
  unsigned short* Vt  = Kh + NK;
  unsigned short* wq  = Vt + NK;
  unsigned short* wk  = wq + NW;
  unsigned short* wv  = wk + NW;
  unsigned short* wo  = wv + NW;
  unsigned short* Xq  = wo + NW;
  unsigned short* Xk  = Xq + NQ;
  unsigned short* Xv  = Xk + NK;
  unsigned short* Ao  = Xq;   // staged Q activations are dead by then

  const float SCALE = 0.125f;  // 1/sqrt(64), folded into Q projection

  cvt_kernel<<<(int)(NQ / 8 / 256), 256, 0, stream>>>(q_in, Xq, (int)(NQ / 8));
  cvt_kernel<<<(int)(NK / 8 / 256), 256, 0, stream>>>(k_in, Xk, (int)(NK / 8));
  cvt_kernel<<<(int)(NK / 8 / 256), 256, 0, stream>>>(v_in, Xv, (int)(NK / 8));
  cvt_kernel<<<(int)(NW / 8 / 256), 256, 0, stream>>>(Wq, wq, (int)(NW / 8));
  cvt_kernel<<<(int)(NW / 8 / 256), 256, 0, stream>>>(Wk, wk, (int)(NW / 8));
  cvt_kernel<<<(int)(NW / 8 / 256), 256, 0, stream>>>(Wv, wv, (int)(NW / 8));
  cvt_kernel<<<(int)(NW / 8 / 256), 256, 0, stream>>>(Wo, wo, (int)(NW / 8));

  int wavesQ  = (BATCH_ * QLEN_ / 16) * 16;    // 4096 wave-tiles
  int wavesKV = (BATCH_ * KVLEN_ / 16) * 16;   // 8192 wave-tiles
  int blocksF = BATCH_ * NH * (QLEN_ / 16) / 8;  // 512 blocks, 8 waves each

  proj_kernel<<<wavesQ / 8, 256, 0, stream>>>(Xq, wq, Qh, 11, 0, SCALE);
  proj_kernel<<<wavesKV / 8, 256, 0, stream>>>(Xk, wk, Kh, 12, 0, 1.0f);
  proj_kernel<<<wavesKV / 8, 256, 0, stream>>>(Xv, wv, Vt, 12, 1, 1.0f);
  flash_kernel<<<blocksF, 256, 0, stream>>>(Qh, Kh, Vt, Ao);
  outproj_kernel<<<wavesQ / 8, 256, 0, stream>>>(Ao, wo, out);
}
